// GNN_81235011436737
// MI455X (gfx1250) — matrix operand order repaired and checked by reading, not compiled, not run on hardware
//
#include <hip/hip_runtime.h>
#include <hip/hip_bf16.h>

// GCNConv on a fixed full graph:  out = ReLU( D^-1/2 (L) D^-1/2 (X W) + b )
// where L is all-ones lower-triangular (upper-tri edges + self loops),
// deg[i] = i+1  ->  agg[i] = dis[i] * prefix_{j<=i}( dis[j] * (XW)[j] ).
// B=512 graphs, N=64 nodes, C=O=256.

#define GB  512
#define GN  64
#define GC  256
#define GO  256
#define LDS_STRIDE 264   // bf16 elements per LDS row (256 + 8 pad, keeps b128 16B-aligned)

typedef __attribute__((ext_vector_type(16))) __bf16 v16bf;
typedef __attribute__((ext_vector_type(8)))  float  v8f;

union FragU {
    uint4  q[2];
    v16bf  b;
};

static __device__ __forceinline__ unsigned short f2bf(float f) {
    // round-to-nearest-even fp32 -> bf16
    unsigned int u = __builtin_bit_cast(unsigned int, f);
    unsigned int rnd = 0x7FFFu + ((u >> 16) & 1u);
    return (unsigned short)((u + rnd) >> 16);
}

// Transpose + convert W (C x O, fp32, row-major) into Wt (O x C, bf16) in workspace.
// Tiny (256 KB read, one-shot per launch); Wt stays hot in the 192 MB L2 for all 512 blocks.
__global__ __launch_bounds__(256) void prep_wt(const float* __restrict__ W,
                                               unsigned short* __restrict__ Wt) {
    const int o = blockIdx.x;      // 256 blocks
    const int c = threadIdx.x;     // 256 threads
    Wt[(size_t)o * GC + c] = f2bf(W[(size_t)c * GO + o]);
}

// One block per batch graph. 256 threads = 8 wave32.
// Wave w computes output column tiles n0 = 2w, n1 = 2w+1 (16 cols each) for all 64 rows.
__global__ __launch_bounds__(256) void gcn_fused(const float* __restrict__ x,
                                                 const unsigned short* __restrict__ Wt,
                                                 const float* __restrict__ bias,
                                                 float* __restrict__ out) {
    __shared__ __align__(16) unsigned short sX[GN * LDS_STRIDE]; // X[b] as bf16 (~33 KB)
    __shared__ float sDis[GN];                                    // dis[i] = rsqrt(i+1)

    const int b   = blockIdx.x;
    const int tid = threadIdx.x;
    const float* __restrict__ xb = x + (size_t)b * GN * GC;

    // ---- Stage 1: X[b] fp32 -> bf16 into LDS (coalesced float4 loads) ----
    {
        const int r  = tid >> 2;          // row 0..63
        const int c0 = (tid & 3) * 64;    // 64-col slab
        const float4* __restrict__ src = (const float4*)(xb + r * GC + c0);
        #pragma unroll
        for (int k = 0; k < 16; ++k) {
            float4 v = src[k];
            unsigned int p0 = (unsigned int)f2bf(v.x) | ((unsigned int)f2bf(v.y) << 16);
            unsigned int p1 = (unsigned int)f2bf(v.z) | ((unsigned int)f2bf(v.w) << 16);
            *(uint2*)&sX[r * LDS_STRIDE + c0 + 4 * k] = make_uint2(p0, p1);
        }
        if (tid < GN) sDis[tid] = rsqrtf((float)(tid + 1));
    }
    __syncthreads();

    const int wave = tid >> 5;
    const int lane = tid & 31;
    const int g    = lane >> 4;   // half-wave (selects K sub-chunks / M sub-rows per ISA layout)
    const int ln   = lane & 15;

    const int nb0 = wave * 32;    // output column bases for this wave's two tiles
    const int nb1 = nb0 + 16;

    v8f acc0[4], acc1[4];
    const v8f vzero = {0.f, 0.f, 0.f, 0.f, 0.f, 0.f, 0.f, 0.f};
    #pragma unroll
    for (int m = 0; m < 4; ++m) { acc0[m] = vzero; acc1[m] = vzero; }

    // ---- Stage 2: XW via bf16 WMMA, fp32 accumulate. K = 256 = 8 x 32 ----
    #pragma unroll
    for (int kb = 0; kb < 8; ++kb) {
        // A fragments: 16-bit A layout -> lane holds row (m*16+ln),
        // K chunks [g*8 .. g*8+7] and [16+g*8 .. 16+g*8+7] (two 16B LDS loads).
        FragU A[4];
        #pragma unroll
        for (int m = 0; m < 4; ++m) {
            const unsigned short* p = &sX[(m * 16 + ln) * LDS_STRIDE + kb * 32 + g * 8];
            A[m].q[0] = *(const uint4*)p;          // K = g*8 ..
            A[m].q[1] = *(const uint4*)(p + 16);   // K = 16 + g*8 ..
        }
        // B fragments: lane = column (nb + ln), K chunks [g*8 .. g*8+7] and [16+g*8 .. 16+g*8+7].
        // Two 16 B loads from the transposed bf16 Wt.
        FragU B0, B1;
        {
            const unsigned short* p = Wt + (size_t)(nb0 + ln) * GC + kb * 32 + g * 8;
            B0.q[0] = *(const uint4*)p;
            B0.q[1] = *(const uint4*)(p + 16);
        }
        {
            const unsigned short* p = Wt + (size_t)(nb1 + ln) * GC + kb * 32 + g * 8;
            B1.q[0] = *(const uint4*)p;
            B1.q[1] = *(const uint4*)(p + 16);
        }
        #pragma unroll
        for (int m = 0; m < 4; ++m) {
            acc0[m] = __builtin_amdgcn_wmma_f32_16x16x32_bf16(
                false, A[m].b, false, B0.b, (short)0, acc0[m], false, false);
            acc1[m] = __builtin_amdgcn_wmma_f32_16x16x32_bf16(
                false, A[m].b, false, B1.b, (short)0, acc1[m], false, false);
        }
    }

    // ---- Stage 3: normalization = in-register prefix scan over 64 rows ----
    // C layout: acc[m][r] holds (row = m*16 + g*8 + r, col = nb + ln).
    // Column is owned by the lane pair (L, L^16); chunk t = 2m + g of 8 rows each.
    float cs0[4], cs1[4];
    #pragma unroll
    for (int m = 0; m < 4; ++m) {
        float r0 = 0.f, r1 = 0.f;
        #pragma unroll
        for (int r = 0; r < 8; ++r) {
            float d = sDis[m * 16 + g * 8 + r];
            r0 += d * acc0[m][r]; acc0[m][r] = r0;   // chunk-local inclusive prefix
            r1 += d * acc1[m][r]; acc1[m][r] = r1;
        }
        cs0[m] = r0; cs1[m] = r1;                    // chunk totals
    }
    // Exchange chunk totals with partner half-wave (same column, other row chunks).
    float ps0[4], ps1[4];
    #pragma unroll
    for (int m = 0; m < 4; ++m) {
        ps0[m] = __shfl_xor(cs0[m], 16, 32);
        ps1[m] = __shfl_xor(cs1[m], 16, 32);
    }
    // Serial offsets over the 8 interleaved chunks (branchless: pair 2m is g=0's chunk).
    float off0[4], off1[4];
    {
        float run0 = 0.f, run1 = 0.f;
        #pragma unroll
        for (int m = 0; m < 4; ++m) {
            off0[m] = run0 + (g ? ps0[m] : 0.f);
            off1[m] = run1 + (g ? ps1[m] : 0.f);
            run0 += cs0[m] + ps0[m];
            run1 += cs1[m] + ps1[m];
        }
    }

    // ---- Stage 4: out = ReLU(dis[i]*(offset + prefix) + bias), coalesced stores ----
    float* __restrict__ ob = out + (size_t)b * GN * GO;
    const float bia0 = bias[nb0 + ln];
    const float bia1 = bias[nb1 + ln];
    #pragma unroll
    for (int m = 0; m < 4; ++m) {
        #pragma unroll
        for (int r = 0; r < 8; ++r) {
            const int row = m * 16 + g * 8 + r;
            const float d = sDis[row];
            float v0 = d * (off0[m] + acc0[m][r]) + bia0;
            float v1 = d * (off1[m] + acc1[m][r]) + bia1;
            ob[row * GO + nb0 + ln] = fmaxf(v0, 0.f);
            ob[row * GO + nb1 + ln] = fmaxf(v1, 0.f);
        }
    }
}

extern "C" void kernel_launch(void* const* d_in, const int* in_sizes, int n_in,
                              void* d_out, int out_size, void* d_ws, size_t ws_size,
                              hipStream_t stream) {
    const float* x    = (const float*)d_in[0];   // [512,64,256] fp32
    const float* W    = (const float*)d_in[1];   // [256,256]    fp32
    const float* bias = (const float*)d_in[2];   // [256]        fp32
    unsigned short* Wt = (unsigned short*)d_ws;  // 256*256 bf16 = 128 KB scratch

    prep_wt<<<dim3(GO), dim3(GC), 0, stream>>>(W, Wt);
    gcn_fused<<<dim3(GB), dim3(256), 0, stream>>>(x, Wt, bias, (float*)d_out);
}
